// SphericalEmbedding_66864050864889
// MI455X (gfx1250) — matrix operand order repaired and checked by reading, not compiled, not run on hardware
//
#include <hip/hip_runtime.h>

#define C_DIM 1536
#define HEADS 16
#define DHEAD 96
#define NOUT  28800
#define NTOT  57600

typedef __attribute__((ext_vector_type(16))) __bf16 v16bf;
typedef __attribute__((ext_vector_type(8)))  float  v8f;

union FragAB { v16bf v; unsigned u[8]; };

__device__ __forceinline__ unsigned short f2bf_rne(float f) {
  unsigned u = __float_as_uint(f);
  unsigned r = u + 0x7FFFu + ((u >> 16) & 1u);
  return (unsigned short)(r >> 16);
}
__device__ __forceinline__ float bf2f(unsigned short s) {
  return __uint_as_float(((unsigned)s) << 16);
}

// ---------------------------------------------------------------------------
// Kernel 1: hf[NTOT][C] = nodes @ W via split-bf16 WMMA (hi*hi + hi*lo + lo*hi)
// Block tile 128(M) x 64(N), 8 waves of wave32, each wave 2x2 16x16 tiles.
// ---------------------------------------------------------------------------
__global__ __launch_bounds__(256)
void sph_gemm_kernel(const float* __restrict__ x, const float* __restrict__ emb,
                     const float* __restrict__ W, float* __restrict__ hf) {
  constexpr int AS = 34;  // ushort stride, A tile 128x32 row-major
  constexpr int BS = 34;  // ushort stride, B tile stored transposed [n][k] 64x32
  __shared__ __align__(16) unsigned short AsH[128 * AS];
  __shared__ __align__(16) unsigned short AsL[128 * AS];
  __shared__ __align__(16) unsigned short BsH[64 * BS];
  __shared__ __align__(16) unsigned short BsL[64 * BS];

  const int tid  = threadIdx.x;
  const int lane = tid & 31;
  const int half = lane >> 4;
  const int l15  = lane & 15;
  const int wave = tid >> 5;
  const int wn   = wave & 1;   // 2 N-subtiles of 32
  const int wm   = wave >> 1;  // 4 M-subtiles of 32
  const int rowBase = blockIdx.y * 128;
  const int colBase = blockIdx.x * 64;
  const bool embHalf = (rowBase >= NOUT);

  v8f acc[2][2] = {};

  for (int k0 = 0; k0 < C_DIM; k0 += 32) {
    // ---- A tile (128 rows x 32 k), bf16 hi/lo split ----
    if (!embHalf) {
      // grid half: nodes[i][k] = x[k*NOUT + i]  (coalesced over i)
#pragma unroll
      for (int rep = 0; rep < 16; ++rep) {
        int idx = rep * 256 + tid;
        int cl = idx >> 7;        // k within tile 0..31
        int il = idx & 127;       // row within tile
        float v = x[(size_t)(k0 + cl) * NOUT + (rowBase + il)];
        unsigned short hi = f2bf_rne(v);
        unsigned short lo = f2bf_rne(v - bf2f(hi));
        AsH[il * AS + cl] = hi;
        AsL[il * AS + cl] = lo;
      }
    } else {
      // emb half: nodes[i][k] = emb[(i-NOUT)*C + k]  (coalesced over k)
#pragma unroll
      for (int rep = 0; rep < 16; ++rep) {
        int idx = rep * 256 + tid;
        int il = idx >> 5;        // row within tile 0..127
        int cl = idx & 31;        // k within tile
        float v = emb[(size_t)(rowBase - NOUT + il) * C_DIM + (k0 + cl)];
        unsigned short hi = f2bf_rne(v);
        unsigned short lo = f2bf_rne(v - bf2f(hi));
        AsH[il * AS + cl] = hi;
        AsL[il * AS + cl] = lo;
      }
    }
    // ---- B tile (32 k x 64 n), stored transposed [n][k] so K-pairs pack ----
#pragma unroll
    for (int rep = 0; rep < 8; ++rep) {
      int idx = rep * 256 + tid;
      int kl = idx >> 6;          // 0..31
      int nl = idx & 63;          // 0..63 (coalesced over n)
      float v = W[(size_t)(k0 + kl) * C_DIM + (colBase + nl)];
      unsigned short hi = f2bf_rne(v);
      unsigned short lo = f2bf_rne(v - bf2f(hi));
      BsH[nl * BS + kl] = hi;
      BsL[nl * BS + kl] = lo;
    }
    __syncthreads();

    // ---- fragments per documented CDNA5 16-bit layouts ----
    FragAB aH[2], aL[2], bH[2], bL[2];
#pragma unroll
    for (int mi = 0; mi < 2; ++mi) {
      int row = wm * 32 + mi * 16 + l15;
#pragma unroll
      for (int i = 0; i < 8; ++i) {
        // A 16x32: lanes0-15 K={0..7,16..23}, lanes16-31 K={8..15,24..31}
        int kk = ((i < 4) ? 0 : 16) + half * 8 + ((i & 3) << 1);
        aH[mi].u[i] = *reinterpret_cast<const unsigned*>(&AsH[row * AS + kk]);
        aL[mi].u[i] = *reinterpret_cast<const unsigned*>(&AsL[row * AS + kk]);
      }
    }
#pragma unroll
    for (int ni = 0; ni < 2; ++ni) {
      int col = wn * 32 + ni * 16 + l15;
#pragma unroll
      for (int i = 0; i < 8; ++i) {
        // B 32x16: lanes0-15 K={0..7,16..23}, lanes16-31 K={8..15,24..31}, col = lane&15
        int kk = ((i < 4) ? 0 : 16) + half * 8 + ((i & 3) << 1);
        bH[ni].u[i] = *reinterpret_cast<const unsigned*>(&BsH[col * BS + kk]);
        bL[ni].u[i] = *reinterpret_cast<const unsigned*>(&BsL[col * BS + kk]);
      }
    }
#pragma unroll
    for (int mi = 0; mi < 2; ++mi)
#pragma unroll
      for (int ni = 0; ni < 2; ++ni) {
        acc[mi][ni] = __builtin_amdgcn_wmma_f32_16x16x32_bf16(
            false, aH[mi].v, false, bH[ni].v, (short)0, acc[mi][ni], false, false);
        acc[mi][ni] = __builtin_amdgcn_wmma_f32_16x16x32_bf16(
            false, aH[mi].v, false, bL[ni].v, (short)0, acc[mi][ni], false, false);
        acc[mi][ni] = __builtin_amdgcn_wmma_f32_16x16x32_bf16(
            false, aL[mi].v, false, bH[ni].v, (short)0, acc[mi][ni], false, false);
      }
    __syncthreads();
  }

  // C/D layout: VGPR r -> M = r + 8*half, N = lane&15
#pragma unroll
  for (int mi = 0; mi < 2; ++mi)
#pragma unroll
    for (int ni = 0; ni < 2; ++ni)
#pragma unroll
      for (int r = 0; r < 8; ++r) {
        int gr = rowBase + wm * 32 + mi * 16 + r + 8 * half;
        int gc = colBase + wn * 32 + ni * 16 + l15;
        hf[(size_t)gr * C_DIM + gc] = acc[mi][ni][r];
      }
}

// ---------------------------------------------------------------------------
// Kernel 2: per-node attention logits a_src/a_dst (N x HEADS)
// ---------------------------------------------------------------------------
__global__ __launch_bounds__(256)
void sph_att_kernel(const float* __restrict__ hf, const float* __restrict__ att_src,
                    const float* __restrict__ att_dst, float* __restrict__ a_src,
                    float* __restrict__ a_dst) {
  __shared__ float sv[C_DIM];
  const int n = blockIdx.x;
  const int tid = threadIdx.x;
#pragma unroll
  for (int j = 0; j < 6; ++j)
    sv[tid + j * 256] = hf[(size_t)n * C_DIM + tid + j * 256];
  __syncthreads();
  if (tid < HEADS) {
    float s = 0.f, d = 0.f;
#pragma unroll 4
    for (int j = 0; j < DHEAD; ++j) {
      float v = sv[tid * DHEAD + j];
      s += v * att_src[tid * DHEAD + j];
      d += v * att_dst[tid * DHEAD + j];
    }
    a_src[n * HEADS + tid] = s;
    a_dst[n * HEADS + tid] = d;
  }
}

// ---------------------------------------------------------------------------
// Kernel 3: self-loop init of denom / agg for output rows (no max-shift needed:
// logits are O(1) so exp()/sum is numerically safe and identical to reference)
// ---------------------------------------------------------------------------
__global__ __launch_bounds__(256)
void sph_selfinit_kernel(const float* __restrict__ hf, const float* __restrict__ a_src,
                         const float* __restrict__ a_dst, float* __restrict__ denom,
                         float* __restrict__ agg) {
  size_t idx = (size_t)blockIdx.x * 256 + threadIdx.x;
  if (idx >= (size_t)NOUT * C_DIM) return;
  int t = (int)(idx / C_DIM);
  int c = (int)(idx - (size_t)t * C_DIM);
  int h = c / DHEAD;
  int g = t + NOUT;
  float e = a_src[g * HEADS + h] + a_dst[g * HEADS + h];
  e = e > 0.f ? e : 0.2f * e;
  float p = expf(e);
  agg[idx] = p * hf[(size_t)g * C_DIM + c];
  if (c == h * DHEAD) denom[t * HEADS + h] = p;
}

// ---------------------------------------------------------------------------
// Kernel 4: graph-edge scatter (irregular dst degree -> float atomics into L2)
// ---------------------------------------------------------------------------
__global__ __launch_bounds__(256)
void sph_scatter_kernel(const long long* __restrict__ edges, int Etot,
                        const float* __restrict__ hf, const float* __restrict__ a_src,
                        const float* __restrict__ a_dst, float* __restrict__ denom,
                        float* __restrict__ agg) {
  __shared__ float p[HEADS];
  const int e = blockIdx.x;                 // only graph edges (< Etot - 2*NOUT)
  const int tid = threadIdx.x;
  const int src = (int)edges[e];
  const int dst = (int)edges[(size_t)Etot + e];
  const int t = dst - NOUT;                 // all graph edges land in upper half
  if (tid < HEADS) {
    float v = a_src[src * HEADS + tid] + a_dst[dst * HEADS + tid];
    v = v > 0.f ? v : 0.2f * v;
    float pp = expf(v);
    p[tid] = pp;
    atomicAdd(&denom[t * HEADS + tid], pp);
  }
  __syncthreads();
#pragma unroll
  for (int j = 0; j < 6; ++j) {
    int c = tid + j * 256;
    atomicAdd(&agg[(size_t)t * C_DIM + c], p[c / DHEAD] * hf[(size_t)src * C_DIM + c]);
  }
}

// ---------------------------------------------------------------------------
// Kernel 5: alpha normalization + bias + LayerNorm, in place in agg
// ---------------------------------------------------------------------------
__global__ __launch_bounds__(256)
void sph_norm_kernel(float* __restrict__ agg, const float* __restrict__ denom,
                     const float* __restrict__ bias, const float* __restrict__ gamma,
                     const float* __restrict__ beta) {
  __shared__ float rs[256];
  __shared__ float rss[256];
  const int r = blockIdx.x;
  const int tid = threadIdx.x;
  float vals[6];
  float s = 0.f, ss = 0.f;
#pragma unroll
  for (int j = 0; j < 6; ++j) {
    int c = tid + j * 256;
    float v = agg[(size_t)r * C_DIM + c] / denom[r * HEADS + c / DHEAD] + bias[c];
    vals[j] = v;
    s += v;
    ss += v * v;
  }
  rs[tid] = s; rss[tid] = ss;
  __syncthreads();
  for (int off = 128; off > 0; off >>= 1) {
    if (tid < off) { rs[tid] += rs[tid + off]; rss[tid] += rss[tid + off]; }
    __syncthreads();
  }
  float mean = rs[0] * (1.0f / C_DIM);
  float var  = rss[0] * (1.0f / C_DIM) - mean * mean;
  float rstd = rsqrtf(var + 1e-5f);
#pragma unroll
  for (int j = 0; j < 6; ++j) {
    int c = tid + j * 256;
    agg[(size_t)r * C_DIM + c] = (vals[j] - mean) * rstd * gamma[c] + beta[c];
  }
}

// ---------------------------------------------------------------------------
// Kernel 6: transpose (NOUT x C) -> (C x NOUT), 32x32 LDS tiles, coalesced both sides
// ---------------------------------------------------------------------------
__global__ __launch_bounds__(256)
void sph_transpose_kernel(const float* __restrict__ agg, float* __restrict__ out) {
  __shared__ float tile[32][33];
  const int bx = blockIdx.x, by = blockIdx.y;
  const int tx = threadIdx.x, ty = threadIdx.y;
#pragma unroll
  for (int j = 0; j < 4; ++j) {
    int r = by * 32 + ty + j * 8;
    int c = bx * 32 + tx;
    tile[ty + j * 8][tx] = agg[(size_t)r * C_DIM + c];
  }
  __syncthreads();
#pragma unroll
  for (int j = 0; j < 4; ++j) {
    int c = bx * 32 + ty + j * 8;
    int r = by * 32 + tx;
    out[(size_t)c * NOUT + r] = tile[tx][ty + j * 8];
  }
}

extern "C" void kernel_launch(void* const* d_in, const int* in_sizes, int n_in,
                              void* d_out, int out_size, void* d_ws, size_t ws_size,
                              hipStream_t stream) {
  const float* x       = (const float*)d_in[0];
  const float* out_emb = (const float*)d_in[1];
  const float* W       = (const float*)d_in[2];
  const float* att_src = (const float*)d_in[3];
  const float* att_dst = (const float*)d_in[4];
  const float* bias    = (const float*)d_in[5];
  const float* gamma   = (const float*)d_in[6];
  const float* beta    = (const float*)d_in[7];
  const long long* edges = (const long long*)d_in[8];

  const int Etot   = in_sizes[8] / 2;       // 201600
  const int nGraph = Etot - 2 * NOUT;       // 144000 (self-loops handled in init)

  float* out = (float*)d_out;
  float* ws  = (float*)d_ws;
  float* hf    = ws;                                   // NTOT*C
  float* a_src = hf    + (size_t)NTOT * C_DIM;         // NTOT*HEADS
  float* a_dst = a_src + (size_t)NTOT * HEADS;         // NTOT*HEADS
  float* denom = a_dst + (size_t)NTOT * HEADS;         // NOUT*HEADS
  float* agg   = denom + (size_t)NOUT * HEADS;         // NOUT*C

  // 1) WMMA GEMM: hf = [x^T ; out_emb] @ W
  sph_gemm_kernel<<<dim3(C_DIM / 64, NTOT / 128), 256, 0, stream>>>(x, out_emb, W, hf);
  // 2) attention logits
  sph_att_kernel<<<NTOT, 256, 0, stream>>>(hf, att_src, att_dst, a_src, a_dst);
  // 3) self-loop init (writes every denom/agg element — no memset needed)
  sph_selfinit_kernel<<<(int)(((size_t)NOUT * C_DIM + 255) / 256), 256, 0, stream>>>(
      hf, a_src, a_dst, denom, agg);
  // 4) graph-edge scatter-aggregate
  sph_scatter_kernel<<<nGraph, 256, 0, stream>>>(edges, Etot, hf, a_src, a_dst, denom, agg);
  // 5) softmax-normalize + bias + LayerNorm (in place)
  sph_norm_kernel<<<NOUT, 256, 0, stream>>>(agg, denom, bias, gamma, beta);
  // 6) transpose to channel-major output
  sph_transpose_kernel<<<dim3(C_DIM / 32, NOUT / 32), dim3(32, 8), 0, stream>>>(agg, out);
}
